// MambaEncoderLayer_72945724555283
// MI455X (gfx1250) — matrix operand order repaired and checked by reading, not compiled, not run on hardware
//
#include <hip/hip_runtime.h>
#include <hip/hip_bf16.h>

// ---------------------------------------------------------------------------
// Mamba encoder layer for gfx1250 (MI455X).  All dense projections run on
// v_wmma_f32_16x16x32_f16 (f16 A/B, f32 accumulate) with 2x2 register
// blocking (32x32 output per wave, 32 B/lane of global traffic per WMMA); the
// selective scan runs as 8192 independent register-resident recurrences.
// ---------------------------------------------------------------------------

#define BATCH   4
#define SEQLEN  2048
#define DMODEL  1024
#define DINNER  2048
#define DSTATE  16
#define DCONV   4
#define DTRANK  64
#define TOKENS  (BATCH * SEQLEN)          // 8192
#define XZCOLS  (2 * DINNER)              // 4096
#define DBCCOLS (DTRANK + 2 * DSTATE)     // 96

typedef __attribute__((ext_vector_type(16))) _Float16 v16h;
typedef __attribute__((ext_vector_type(8)))  _Float16 v8h;
typedef __attribute__((ext_vector_type(8)))  float    v8f;

// ---------------------------------------------------------------------------
// Weight conversion: W[K][N] fp32  ->  Wt[N][K] f16 (transposed so WMMA B
// fragments are contiguous per lane).
// ---------------------------------------------------------------------------
__global__ void cvt_transpose_w(const float* __restrict__ W,
                                _Float16* __restrict__ Wt, int K, int N) {
    long idx = (long)blockIdx.x * blockDim.x + threadIdx.x;
    if (idx >= (long)K * N) return;
    int k = (int)(idx / N);
    int n = (int)(idx % N);
    Wt[(long)n * K + k] = (_Float16)W[(long)k * N + n];
}

// ---------------------------------------------------------------------------
// LayerNorm: one 256-thread block per row, LDS tree reduction; emits f16
// activations (WMMA A operand).
// ---------------------------------------------------------------------------
__global__ __launch_bounds__(256)
void layernorm_f16(const float* __restrict__ X, const float* __restrict__ g,
                   const float* __restrict__ bta, _Float16* __restrict__ Y, int D) {
    __shared__ float red[256];
    const int row = blockIdx.x;
    const float* x = X + (long)row * D;
    float s = 0.f;
    for (int i = threadIdx.x; i < D; i += 256) s += x[i];
    red[threadIdx.x] = s; __syncthreads();
    for (int off = 128; off > 0; off >>= 1) {
        if (threadIdx.x < off) red[threadIdx.x] += red[threadIdx.x + off];
        __syncthreads();
    }
    const float mu = red[0] / D;
    __syncthreads();
    float v = 0.f;
    for (int i = threadIdx.x; i < D; i += 256) { float d = x[i] - mu; v += d * d; }
    red[threadIdx.x] = v; __syncthreads();
    for (int off = 128; off > 0; off >>= 1) {
        if (threadIdx.x < off) red[threadIdx.x] += red[threadIdx.x + off];
        __syncthreads();
    }
    const float rstd = rsqrtf(red[0] / D + 1e-5f);
    for (int i = threadIdx.x; i < D; i += 256)
        Y[(long)row * D + i] = (_Float16)((x[i] - mu) * rstd * g[i] + bta[i]);
}

// ---------------------------------------------------------------------------
// WMMA fragment helpers (16-bit A 16x32 / B 32x16 layouts as used here, k = 16*(l>=16) + elem).
// A: lane l holds row m0+(l&15); elems 0..15 = K[16*(l>=16) .. +15]
//    -- one contiguous 32 B load, same k order as B.
// B (from transposed weight [N][K]): lane l holds col n0+(l&15),
//    elems 0..15 = K[16*(l>=16) .. +15] -- one contiguous 32 B load.
// ---------------------------------------------------------------------------
__device__ __forceinline__ v16h load_afrag(const _Float16* p) {
    v8h lo = *(const v8h*)p;
    v8h hi = *(const v8h*)(p + 16);
    v16h r;
#pragma unroll
    for (int i = 0; i < 8; ++i) { r[i] = lo[i]; r[i + 8] = hi[i]; }
    return r;
}

__device__ __forceinline__ v8f wmma_f16(v16h a, v16h b, v8f c) {
    return __builtin_amdgcn_wmma_f32_16x16x32_f16(
        /*neg_a=*/false, a, /*neg_b=*/false, b,
        /*c_mod=*/(short)0, c, /*reuse_a=*/false, /*reuse_b=*/false);
}

__device__ __forceinline__ void store_ctile(float* __restrict__ C, int N,
                                            int m0, int n0, int hi, int l15, v8f c) {
    float* crow = C + (long)(m0 + hi * 8) * N + n0 + l15;
#pragma unroll
    for (int r = 0; r < 8; ++r) crow[(long)r * N] = c[r];
}

// ---------------------------------------------------------------------------
// WMMA GEMM: C[M][N] (f32) = A[M][K] (f16, row-major) * Bt[N][K] (f16).
// Block = 4 waves covering a 64x64 output tile; each wave computes a 32x32
// block as 2x2 register subtiles (4 v8f accumulators).  Per K-step: 2 A
// fragments + 2 B fragments feed 4 WMMAs (32 B/lane global traffic per WMMA).
// All control flow is wave-uniform so EXEC is all-ones at every WMMA.
// ---------------------------------------------------------------------------
__global__ __launch_bounds__(128)
void wmma_gemm_f16f32(const _Float16* __restrict__ A,
                      const _Float16* __restrict__ Bt,
                      float* __restrict__ C, int M, int N, int K) {
    const int lane = threadIdx.x & 31;
    const int wave = threadIdx.x >> 5;
    const int mw = wave >> 1;            // 0/1: m half of block tile
    const int nw = wave & 1;             // 0/1: n half of block tile
    const int m0 = blockIdx.y * 64 + mw * 32;
    const int n0 = blockIdx.x * 64 + nw * 32;
    const int hi  = lane >> 4;
    const int l15 = lane & 15;

    const bool nv1 = (n0 + 16 < N);      // second n subtile valid (wave-uniform)
    if (n0 >= N) return;                 // wave-uniform

    const _Float16* arow0 = A  + (long)(m0 + l15) * K + (hi ? 16 : 0);
    const _Float16* arow1 = arow0 + (long)16 * K;
    const _Float16* brow0 = Bt + (long)(n0 + l15) * K + hi * 16;
    const _Float16* brow1 = brow0 + (long)16 * K;

    v8f c00 = {}, c01 = {}, c10 = {}, c11 = {};
    const int PF = 512;                  // prefetch distance: 1 KB of halves
    for (int k0 = 0; k0 < K; k0 += 32) {
        if (k0 + PF < K) {               // uniform; lowers to global_prefetch_b8
            __builtin_prefetch(arow0 + k0 + PF, 0, 1);
            __builtin_prefetch(brow0 + k0 + PF, 0, 1);
        }
        v16h a0 = *(const v16h*)(arow0 + k0);
        v16h a1 = *(const v16h*)(arow1 + k0);
        v16h b0 = *(const v16h*)(brow0 + k0);
        c00 = wmma_f16(a0, b0, c00);
        c10 = wmma_f16(a1, b0, c10);
        if (nv1) {
            v16h b1 = *(const v16h*)(brow1 + k0);
            c01 = wmma_f16(a0, b1, c01);
            c11 = wmma_f16(a1, b1, c11);
        }
    }

    store_ctile(C, N, m0,      n0,      hi, l15, c00);
    store_ctile(C, N, m0 + 16, n0,      hi, l15, c10);
    if (nv1) {
        store_ctile(C, N, m0,      n0 + 16, hi, l15, c01);
        store_ctile(C, N, m0 + 16, n0 + 16, hi, l15, c11);
    }
}

// ---------------------------------------------------------------------------
// Depthwise causal conv (taps=4) + bias + SiLU over the u-half of xz.
// ---------------------------------------------------------------------------
__global__ void conv_silu(const float* __restrict__ xz,
                          const float* __restrict__ cw,
                          const float* __restrict__ cb,
                          float* __restrict__ u32, _Float16* __restrict__ u16) {
    long idx = (long)blockIdx.x * blockDim.x + threadIdx.x;
    if (idx >= (long)TOKENS * DINNER) return;
    int  d = (int)(idx % DINNER);
    long t = idx / DINNER;
    int  l = (int)(t % SEQLEN);
    float acc = cb[d];
#pragma unroll
    for (int k = 0; k < DCONV; ++k) {
        int ls = l + k - (DCONV - 1);
        if (ls >= 0)
            acc += xz[(t + (k - (DCONV - 1))) * XZCOLS + d] * cw[d * DCONV + k];
    }
    float s = acc / (1.f + __expf(-acc));   // SiLU
    u32[idx] = s;
    u16[idx] = (_Float16)s;
}

// dt = dbc[:, :64] as f16 (A operand of dt_proj GEMM)
__global__ void extract_dt(const float* __restrict__ dbc, _Float16* __restrict__ dt) {
    long idx = (long)blockIdx.x * blockDim.x + threadIdx.x;
    if (idx >= (long)TOKENS * DTRANK) return;
    long t = idx / DTRANK;
    int  r = (int)(idx % DTRANK);
    dt[idx] = (_Float16)dbc[t * DBCCOLS + r];
}

// delta = softplus(dtp + dt_proj_b), in place
__global__ void softplus_bias(float* __restrict__ dtp, const float* __restrict__ bias) {
    long idx = (long)blockIdx.x * blockDim.x + threadIdx.x;
    if (idx >= (long)TOKENS * DINNER) return;
    int d = (int)(idx % DINNER);
    float x = dtp[idx] + bias[d];
    dtp[idx] = (x > 20.f) ? x : log1pf(__expf(x));
}

// ---------------------------------------------------------------------------
// Selective scan: one thread per (b,d) chain; 16-wide state in registers.
// B/C loads are wave-uniform broadcasts (all lanes share (b,l)).
// ---------------------------------------------------------------------------
__global__ __launch_bounds__(256)
void scan_kernel(const float* __restrict__ u, const float* __restrict__ delta,
                 const float* __restrict__ dbc, const float* __restrict__ A_log,
                 const float* __restrict__ Dp, float* __restrict__ y) {
    int idx = blockIdx.x * blockDim.x + threadIdx.x;
    if (idx >= BATCH * DINNER) return;
    const int b = idx / DINNER;
    const int d = idx % DINNER;

    float Ar[DSTATE];
#pragma unroll
    for (int s = 0; s < DSTATE; ++s) Ar[s] = -__expf(A_log[d * DSTATE + s]);
    const float Dd = Dp[d];

    float h[DSTATE];
#pragma unroll
    for (int s = 0; s < DSTATE; ++s) h[s] = 0.f;

    for (int l = 0; l < SEQLEN; ++l) {
        const long  t  = (long)b * SEQLEN + l;
        const float uv = u[t * DINNER + d];
        const float dl = delta[t * DINNER + d];
        const float du = dl * uv;
        const float* Bm = dbc + t * DBCCOLS + DTRANK;
        const float* Cm = Bm + DSTATE;
        float acc = uv * Dd;
#pragma unroll
        for (int s = 0; s < DSTATE; ++s) {
            h[s] = __expf(dl * Ar[s]) * h[s] + du * Bm[s];
            acc += h[s] * Cm[s];
        }
        y[t * DINNER + d] = acc;
    }
}

// y <- y * silu(z), emitted as f16 (A operand of out_proj GEMM)
__global__ void gate_kernel(const float* __restrict__ y, const float* __restrict__ xz,
                            _Float16* __restrict__ y16) {
    long idx = (long)blockIdx.x * blockDim.x + threadIdx.x;
    if (idx >= (long)TOKENS * DINNER) return;
    int  d = (int)(idx % DINNER);
    long t = idx / DINNER;
    float z = xz[t * XZCOLS + DINNER + d];
    float g = z / (1.f + __expf(-z));
    y16[idx] = (_Float16)(y[idx] * g);
}

// out = gelu(acc + bias), exact gelu via erf
__global__ void bias_gelu(const float* __restrict__ acc, const float* __restrict__ bias,
                          float* __restrict__ out) {
    long idx = (long)blockIdx.x * blockDim.x + threadIdx.x;
    if (idx >= (long)TOKENS * DMODEL) return;
    int d = (int)(idx % DMODEL);
    float x = acc[idx] + bias[d];
    out[idx] = 0.5f * x * (1.f + erff(x * 0.70710678118654752f));
}

// ---------------------------------------------------------------------------
// Host-side orchestration
// ---------------------------------------------------------------------------
static inline int ceil_div(long a, long b) { return (int)((a + b - 1) / b); }

extern "C" void kernel_launch(void* const* d_in, const int* in_sizes, int n_in,
                              void* d_out, int out_size, void* d_ws, size_t ws_size,
                              hipStream_t stream) {
    const float* x         = (const float*)d_in[0];
    const float* ln1_g     = (const float*)d_in[1];
    const float* ln1_b     = (const float*)d_in[2];
    const float* ln2_g     = (const float*)d_in[3];
    const float* ln2_b     = (const float*)d_in[4];
    const float* in_proj_w = (const float*)d_in[5];
    const float* conv_w    = (const float*)d_in[6];
    const float* conv_b    = (const float*)d_in[7];
    const float* x_proj_w  = (const float*)d_in[8];
    const float* dt_proj_w = (const float*)d_in[9];
    const float* dt_proj_b = (const float*)d_in[10];
    const float* A_log     = (const float*)d_in[11];
    const float* Dp        = (const float*)d_in[12];
    const float* out_proj_w= (const float*)d_in[13];
    const float* mlp_w     = (const float*)d_in[14];
    const float* mlp_b     = (const float*)d_in[15];
    float* out = (float*)d_out;

    // workspace carve-up (256B aligned)
    char* p = (char*)d_ws;
    auto carve = [&](size_t bytes) -> void* {
        void* r = (void*)p;
        p += (bytes + 255) & ~(size_t)255;
        return r;
    };
    _Float16* wt_in  = (_Float16*)carve((size_t)XZCOLS * DMODEL * 2);  // [4096][1024]
    _Float16* wt_xp  = (_Float16*)carve((size_t)DBCCOLS * DINNER * 2); // [96][2048]
    _Float16* wt_dt  = (_Float16*)carve((size_t)DINNER * DTRANK * 2);  // [2048][64]
    _Float16* wt_out = (_Float16*)carve((size_t)DMODEL * DINNER * 2);  // [1024][2048]
    _Float16* wt_mlp = (_Float16*)carve((size_t)DMODEL * DMODEL * 2);  // [1024][1024]
    _Float16* h1f16  = (_Float16*)carve((size_t)TOKENS * DMODEL * 2);
    float*    xz     = (float*)   carve((size_t)TOKENS * XZCOLS * 4);
    float*    u32    = (float*)   carve((size_t)TOKENS * DINNER * 4);
    _Float16* u16    = (_Float16*)carve((size_t)TOKENS * DINNER * 2);
    float*    dbc    = (float*)   carve((size_t)TOKENS * DBCCOLS * 4);
    _Float16* dt16   = (_Float16*)carve((size_t)TOKENS * DTRANK * 2);
    float*    delta  = (float*)   carve((size_t)TOKENS * DINNER * 4);
    float*    ybuf   = (float*)   carve((size_t)TOKENS * DINNER * 4);
    _Float16* y16    = (_Float16*)carve((size_t)TOKENS * DINNER * 2);
    float*    mout   = (float*)   carve((size_t)TOKENS * DMODEL * 4);
    _Float16* h2f16  = (_Float16*)carve((size_t)TOKENS * DMODEL * 2);
    float*    mlpraw = (float*)   carve((size_t)TOKENS * DMODEL * 4);
    (void)ws_size; (void)n_in; (void)in_sizes; (void)out_size;

    const int EW = 256;

    // 1) convert + transpose all weights to f16 [N][K]
    cvt_transpose_w<<<ceil_div((long)DMODEL * XZCOLS, EW), EW, 0, stream>>>(in_proj_w,  wt_in,  DMODEL, XZCOLS);
    cvt_transpose_w<<<ceil_div((long)DINNER * DBCCOLS, EW), EW, 0, stream>>>(x_proj_w,  wt_xp,  DINNER, DBCCOLS);
    cvt_transpose_w<<<ceil_div((long)DTRANK * DINNER, EW), EW, 0, stream>>>(dt_proj_w,  wt_dt,  DTRANK, DINNER);
    cvt_transpose_w<<<ceil_div((long)DINNER * DMODEL, EW), EW, 0, stream>>>(out_proj_w, wt_out, DINNER, DMODEL);
    cvt_transpose_w<<<ceil_div((long)DMODEL * DMODEL, EW), EW, 0, stream>>>(mlp_w,      wt_mlp, DMODEL, DMODEL);

    // 2) LN1 -> f16
    layernorm_f16<<<TOKENS, 256, 0, stream>>>(x, ln1_g, ln1_b, h1f16, DMODEL);

    // 3) xz = h1 @ in_proj  (8192 x 1024 x 4096)
    {
        dim3 grid(XZCOLS / 64, TOKENS / 64);
        wmma_gemm_f16f32<<<grid, 128, 0, stream>>>(h1f16, wt_in, xz, TOKENS, XZCOLS, DMODEL);
    }

    // 4) conv + SiLU
    conv_silu<<<ceil_div((long)TOKENS * DINNER, EW), EW, 0, stream>>>(xz, conv_w, conv_b, u32, u16);

    // 5) dbc = u @ x_proj  (8192 x 2048 x 96)
    {
        dim3 grid(ceil_div(DBCCOLS, 64), TOKENS / 64);
        wmma_gemm_f16f32<<<grid, 128, 0, stream>>>(u16, wt_xp, dbc, TOKENS, DBCCOLS, DINNER);
    }

    // 6) delta = softplus(dt @ dt_proj + b)  (8192 x 64 x 2048)
    extract_dt<<<ceil_div((long)TOKENS * DTRANK, EW), EW, 0, stream>>>(dbc, dt16);
    {
        dim3 grid(DINNER / 64, TOKENS / 64);
        wmma_gemm_f16f32<<<grid, 128, 0, stream>>>(dt16, wt_dt, delta, TOKENS, DINNER, DTRANK);
    }
    softplus_bias<<<ceil_div((long)TOKENS * DINNER, EW), EW, 0, stream>>>(delta, dt_proj_b);

    // 7) selective scan (+ u*Dp)
    scan_kernel<<<ceil_div((long)BATCH * DINNER, EW), EW, 0, stream>>>(u32, delta, dbc, A_log, Dp, ybuf);

    // 8) gate: y *= silu(z), emit f16
    gate_kernel<<<ceil_div((long)TOKENS * DINNER, EW), EW, 0, stream>>>(ybuf, xz, y16);

    // 9) mamba_out = y @ out_proj  (8192 x 2048 x 1024)
    {
        dim3 grid(DMODEL / 64, TOKENS / 64);
        wmma_gemm_f16f32<<<grid, 128, 0, stream>>>(y16, wt_out, mout, TOKENS, DMODEL, DINNER);
    }

    // 10) LN2 -> f16
    layernorm_f16<<<TOKENS, 256, 0, stream>>>(mout, ln2_g, ln2_b, h2f16, DMODEL);

    // 11) mlp_raw = h2 @ mlp_w  (8192 x 1024 x 1024)
    {
        dim3 grid(DMODEL / 64, TOKENS / 64);
        wmma_gemm_f16f32<<<grid, 128, 0, stream>>>(h2f16, wt_mlp, mlpraw, TOKENS, DMODEL, DMODEL);
    }

    // 12) out = gelu(mlp_raw + mlp_b)
    bias_gelu<<<ceil_div((long)TOKENS * DMODEL, EW), EW, 0, stream>>>(mlpraw, mlp_b, out);
}
